// Attention_52055003628266
// MI455X (gfx1250) — matrix operand order repaired and checked by reading, not compiled, not run on hardware
//
#include <hip/hip_runtime.h>
#include <hip/hip_bf16.h>
#include <math.h>

// ---------------- problem constants ----------------
#define SEQ       196
#define DIM       768
#define HEADS     12
#define HEAD_DIM  64
#define BATCH     64
#define ROWS      (BATCH * SEQ)   // 12544
#define QKV_N     (3 * DIM)       // 2304
#define NP16      208             // SEQ padded to multiple of 16
#define NP32      224             // SEQ padded to multiple of 32 (K of P@V)
#define SCALE_F   0.125f          // 64^-0.5

// ---------------- WMMA fragment types ----------------
typedef __attribute__((ext_vector_type(16))) __bf16 v16bf;
typedef __attribute__((ext_vector_type(8)))  __bf16 v8bf;
typedef __attribute__((ext_vector_type(8)))  float  v8f;

union Frag { v16bf v; v8bf h[2]; };

static __device__ inline v8bf bf8_zero() {
    v8bf z;
#pragma unroll
    for (int i = 0; i < 8; ++i) z[i] = (__bf16)0.0f;
    return z;
}

// D = A(16x32) * B(32x16) + C, bf16 in / f32 accum
static __device__ inline v8f wmma_bf16(const Frag& a, const Frag& b, v8f c) {
    return __builtin_amdgcn_wmma_f32_16x16x32_bf16(
        false, a.v, false, b.v, (short)0, c, false, false);
}

// LDS byte-offset of a generic pointer to __shared__ memory:
// flat LDS aperture address maps to LDS via addr[31:0] (ISA 10.2).
static __device__ inline unsigned lds_off(const void* p) {
    return (unsigned)(uintptr_t)p;
}

// async global -> LDS copy of 16 bytes per lane (ASYNCcnt-tracked).
// GVS addressing: mem = SGPR_base + VGPR_voffset(bytes) ; LDS dst offset in VGPR.
static __device__ inline void async_g2l_b128(unsigned lds_byte_off,
                                             unsigned voff_bytes,
                                             const void* sbase) {
    asm volatile("global_load_async_to_lds_b128 %0, %1, %2"
                 :: "v"(lds_byte_off), "v"(voff_bytes), "s"(sbase)
                 : "memory");
}

static __device__ inline void wait_async0() {
    asm volatile("s_wait_asynccnt 0x0" ::: "memory");
}

// ---------------- conversion kernels ----------------
__global__ void cvt_f32_to_bf16(const float* __restrict__ src,
                                __bf16* __restrict__ dst, int n) {
    int i = blockIdx.x * blockDim.x + threadIdx.x;
    int stride = gridDim.x * blockDim.x;
    for (; i < n; i += stride) dst[i] = (__bf16)src[i];
}

// src [K x N] row-major f32  ->  dst [N x K] row-major bf16 (transpose)
__global__ void cvt_transpose_bf16(const float* __restrict__ src,
                                   __bf16* __restrict__ dst, int K, int N) {
    int i = blockIdx.x * blockDim.x + threadIdx.x;
    int stride = gridDim.x * blockDim.x;
    int total = K * N;
    for (; i < total; i += stride) {
        int k = i / N, n = i - k * N;
        dst[(size_t)n * K + k] = (__bf16)src[i];
    }
}

// ---------------- block-tiled WMMA GEMM with async-LDS staging ----------------
// A [M x K] bf16 row-major, Bt [N x K] bf16 row-major (B transposed).
// 256 threads / 8 waves per block; block tile 128x128; wave tile 32x64.
// K staged 64-deep, double-buffered in LDS, filled with
// global_load_async_to_lds_b128 and consumed as ds_load_b128 fragments.
// Requires M%128==0, N%128==0, K%64==0 (true for all uses here).
//
// Per-lane fragment layouts (ISA 7.12.2):
//   A: row = lane&15 ; 8-elem chunks at kk+(lane>>4)*8 and kk+16+(lane>>4)*8
//   B: col = lane&15 ; 8-elem chunks at kk+(lane>>4)*8 and kk+16+(lane>>4)*8
//   C/D: VGPR j -> row (j + 8*(lane>>4)), col (lane&15)

#define BT_M 128
#define BT_N 128
#define BT_K 64

template <bool OUT_BF16>
__global__ __launch_bounds__(256)
void gemm_wmma(const __bf16* __restrict__ A, const __bf16* __restrict__ Bt,
               const float* __restrict__ bias, void* __restrict__ Cout,
               int M, int N, int K) {
    __shared__ __align__(16) __bf16 Abuf[2][BT_M * BT_K];   // 2 x 16 KB
    __shared__ __align__(16) __bf16 Bbuf[2][BT_N * BT_K];   // 2 x 16 KB

    const int tid  = threadIdx.x;
    const int lane = tid & 31, wave = tid >> 5;
    const int r    = lane & 15, hf = lane >> 4;
    const int wm   = wave >> 1;          // 0..3  (rows of wave grid)
    const int wn   = wave & 1;           // 0..1  (cols of wave grid)
    const int m0   = blockIdx.y * BT_M;
    const int n0   = blockIdx.x * BT_N;

    // Per-thread copy assignments: a stage is 1024 16-byte units per matrix
    // (128 rows x 64 cols x 2B); each thread moves 4 units of A and 4 of B.
    // Unit u -> row u>>3, elem-col (u&7)*8 ; LDS layout row-major [row][64].
    unsigned voffA[4], voffB[4], uoff[4];
    const unsigned baseA = lds_off(&Abuf[0][0]);
    const unsigned baseB = lds_off(&Bbuf[0][0]);
    const unsigned bufBytes = BT_M * BT_K * 2;
#pragma unroll
    for (int i = 0; i < 4; ++i) {
        unsigned u = (unsigned)tid + 256u * i;
        unsigned row = u >> 3, c = (u & 7) * 8;
        voffA[i] = ((unsigned)(m0 + row) * (unsigned)K + c) * 2u;
        voffB[i] = ((unsigned)(n0 + row) * (unsigned)K + c) * 2u;
        uoff[i]  = u * 16u;
    }

    v8f acc[2][4] = {};
    const int NSTAGE = K / BT_K;

    // prologue: fill buffer 0
#pragma unroll
    for (int i = 0; i < 4; ++i) {
        async_g2l_b128(baseA + uoff[i], voffA[i], A);
        async_g2l_b128(baseB + uoff[i], voffB[i], Bt);
    }
    wait_async0();
    __syncthreads();

    for (int s = 0; s < NSTAGE; ++s) {
        const int cur = s & 1;
        if (s + 1 < NSTAGE) {
            const __bf16* sA = A  + (s + 1) * BT_K;
            const __bf16* sB = Bt + (s + 1) * BT_K;
            const unsigned dst = (unsigned)(cur ^ 1) * bufBytes;
#pragma unroll
            for (int i = 0; i < 4; ++i) {
                async_g2l_b128(baseA + dst + uoff[i], voffA[i], sA);
                async_g2l_b128(baseB + dst + uoff[i], voffB[i], sB);
            }
        }

        const __bf16* Ab = &Abuf[cur][0];
        const __bf16* Bb = &Bbuf[cur][0];
#pragma unroll
        for (int kk = 0; kk < BT_K; kk += 32) {
            Frag a[2], b[4];
#pragma unroll
            for (int mt = 0; mt < 2; ++mt) {
                const __bf16* p = Ab + (wm * 32 + mt * 16 + r) * BT_K + kk + hf * 8;
                a[mt].h[0] = *(const v8bf*)p;
                a[mt].h[1] = *(const v8bf*)(p + 16);
            }
#pragma unroll
            for (int nt = 0; nt < 4; ++nt) {
                const __bf16* p = Bb + (wn * 64 + nt * 16 + r) * BT_K + kk + hf * 8;
                b[nt].h[0] = *(const v8bf*)p;
                b[nt].h[1] = *(const v8bf*)(p + 16);
            }
#pragma unroll
            for (int mt = 0; mt < 2; ++mt)
#pragma unroll
                for (int nt = 0; nt < 4; ++nt)
                    acc[mt][nt] = wmma_bf16(a[mt], b[nt], acc[mt][nt]);
        }

        wait_async0();       // next-stage fill (this wave's share) landed
        __syncthreads();     // all waves' shares landed / done reading
    }

#pragma unroll
    for (int mt = 0; mt < 2; ++mt)
#pragma unroll
        for (int nt = 0; nt < 4; ++nt)
#pragma unroll
            for (int j = 0; j < 8; ++j) {
                int row = m0 + wm * 32 + mt * 16 + j + 8 * hf;
                int col = n0 + wn * 64 + nt * 16 + r;
                if constexpr (OUT_BF16)
                    ((__bf16*)Cout)[(size_t)row * N + col] = (__bf16)acc[mt][nt][j];
                else
                    ((float*)Cout)[(size_t)row * N + col] = acc[mt][nt][j] + bias[col];
            }
}

// ---------------- fused attention (per b,h workgroup) ----------------
// LDS layout (bytes):
//   qs : [NP16 x 64]  bf16   @ 0       (26624)
//   ks : [NP16 x 64]  bf16   @ 26624   (26624)
//   vT : [64 x NP32]  bf16   @ 53248   (28672)
//   S  : [NP16 x NP32] f32   @ 81920   (186368)  -- reused in-place as bf16 P
// total 268288 B  (<= 320 KB/WGP)
#define LDS_Q  0
#define LDS_K  26624
#define LDS_VT 53248
#define LDS_S  81920
#define LDS_TOTAL 268288

__global__ __launch_bounds__(256)
void attn_kernel(const __bf16* __restrict__ qkv,      // [ROWS x 2304]
                 const float*  __restrict__ static_a, // [1,H,SEQ,SEQ]
                 __bf16*       __restrict__ attn_out) // [ROWS x 768]
{
    extern __shared__ __align__(16) char smem[];
    __bf16* qs = (__bf16*)(smem + LDS_Q);
    __bf16* ks = (__bf16*)(smem + LDS_K);
    __bf16* vT = (__bf16*)(smem + LDS_VT);
    float*  S  = (float*) (smem + LDS_S);   // row stride NP32 floats
    __bf16* P  = (__bf16*)(smem + LDS_S);   // row stride 2*NP32 bf16 elems

    const int bh  = blockIdx.x;
    const int b   = bh / HEADS;
    const int h   = bh - b * HEADS;
    const int tid = threadIdx.x;
    const size_t base = ((size_t)b * SEQ) * QKV_N + (size_t)h * HEAD_DIM;

    // ---- stage q, k (row-major, padded rows zeroed) ----
    for (int i = tid; i < NP16 * 8; i += 256) {
        int n = i >> 3, c = i & 7;
        if (n < SEQ) {
            const __bf16* src = qkv + base + (size_t)n * QKV_N + c * 8;
            *(v8bf*)(qs + n * 64 + c * 8) = *(const v8bf*)(src);        // q
            *(v8bf*)(ks + n * 64 + c * 8) = *(const v8bf*)(src + DIM);  // k
        } else {
            v8bf z = bf8_zero();
            *(v8bf*)(qs + n * 64 + c * 8) = z;
            *(v8bf*)(ks + n * 64 + c * 8) = z;
        }
    }
    // ---- stage v transposed: vT[d][m], padded m zeroed ----
    for (int i = tid; i < 64 * NP32; i += 256) {
        int d = i / NP32, m = i - d * NP32;
        __bf16 val = (__bf16)0.0f;
        if (m < SEQ) val = qkv[base + (size_t)m * QKV_N + 2 * DIM + d];
        vT[d * NP32 + m] = val;
    }
    __syncthreads();

    const int lane = tid & 31, wave = tid >> 5;
    const int r = lane & 15, hf = lane >> 4;

    // ---- S = q @ k^T  (13x13 tiles of 16x16, K=64) ----
    for (int t = wave; t < 169; t += 8) {
        int n0 = (t / 13) * 16, m0 = (t % 13) * 16;
        v8f acc = {};
#pragma unroll
        for (int kb = 0; kb < 64; kb += 32) {
            Frag a, bb;
            const __bf16* pa = qs + (n0 + r) * 64 + kb + hf * 8;
            a.h[0] = *(const v8bf*)pa;
            a.h[1] = *(const v8bf*)(pa + 16);
            const __bf16* pb = ks + (m0 + r) * 64 + kb + hf * 8;
            bb.h[0] = *(const v8bf*)pb;
            bb.h[1] = *(const v8bf*)(pb + 16);
            acc = wmma_bf16(a, bb, acc);
        }
#pragma unroll
        for (int j = 0; j < 8; ++j)
            S[(n0 + j + 8 * hf) * NP32 + m0 + r] = acc[j];
    }
    __syncthreads();

    // ---- per-row softmax (+ post-softmax bias), write bf16 P in place ----
    for (int n = wave; n < NP16; n += 8) {
        __bf16* prow = P + n * (2 * NP32);
        if (n >= SEQ) {
#pragma unroll
            for (int j = 0; j < 7; ++j) prow[lane + 32 * j] = (__bf16)0.0f;
            continue;
        }
        float sv[7];
#pragma unroll
        for (int j = 0; j < 7; ++j) {
            int m = lane + 32 * j;
            sv[j] = (m < SEQ) ? S[n * NP32 + m] * SCALE_F : -INFINITY;
        }
        float mx = sv[0];
#pragma unroll
        for (int j = 1; j < 7; ++j) mx = fmaxf(mx, sv[j]);
        for (int off = 16; off >= 1; off >>= 1)
            mx = fmaxf(mx, __shfl_xor(mx, off, 32));
        float ev[7], sum = 0.0f;
#pragma unroll
        for (int j = 0; j < 7; ++j) { ev[j] = __expf(sv[j] - mx); sum += ev[j]; }
        for (int off = 16; off >= 1; off >>= 1)
            sum += __shfl_xor(sum, off, 32);
        float inv = 1.0f / sum;
        const float* ab = static_a + ((size_t)h * SEQ + n) * SEQ;
#pragma unroll
        for (int j = 0; j < 7; ++j) {
            int m = lane + 32 * j;
            float pv = 0.0f;
            if (m < SEQ) pv = ev[j] * inv + ab[m];   // softmax + additive bias
            prow[m] = (__bf16)pv;                    // padded cols -> 0
        }
    }
    __syncthreads();

    // ---- out = P @ v  (13 n-tiles x 4 d-tiles, K = NP32 = 224) ----
    for (int t = wave; t < 52; t += 8) {
        int n0 = (t / 4) * 16, d0 = (t % 4) * 16;
        v8f acc = {};
#pragma unroll
        for (int kb = 0; kb < NP32; kb += 32) {
            Frag a, bb;
            const __bf16* pa = P + (n0 + r) * (2 * NP32) + kb + hf * 8;
            a.h[0] = *(const v8bf*)pa;
            a.h[1] = *(const v8bf*)(pa + 16);
            const __bf16* pb = vT + (d0 + r) * NP32 + kb + hf * 8;
            bb.h[0] = *(const v8bf*)pb;
            bb.h[1] = *(const v8bf*)(pb + 16);
            acc = wmma_bf16(a, bb, acc);
        }
#pragma unroll
        for (int j = 0; j < 8; ++j) {
            int n = n0 + j + 8 * hf;
            if (n < SEQ)
                attn_out[((size_t)b * SEQ + n) * DIM + h * HEAD_DIM + d0 + r] =
                    (__bf16)acc[j];
        }
    }
}

// ---------------- host launcher ----------------
extern "C" void kernel_launch(void* const* d_in, const int* in_sizes, int n_in,
                              void* d_out, int out_size, void* d_ws, size_t ws_size,
                              hipStream_t stream) {
    const float* x        = (const float*)d_in[0];
    const float* W_qkv    = (const float*)d_in[1];
    const float* static_a = (const float*)d_in[2];
    const float* W_proj   = (const float*)d_in[3];
    const float* b_proj   = (const float*)d_in[4];
    float* out = (float*)d_out;

    // workspace carve-up (256B aligned), ~96.4 MB total
    char* ws = (char*)d_ws;
    size_t off = 0;
    auto carve = [&](size_t bytes) -> void* {
        void* p = ws + off;
        off = (off + bytes + 255) & ~(size_t)255;
        return p;
    };
    __bf16* xb      = (__bf16*)carve((size_t)ROWS * DIM * 2);     // x in bf16
    __bf16* wqkvT   = (__bf16*)carve((size_t)QKV_N * DIM * 2);    // W_qkv^T bf16
    __bf16* wprojT  = (__bf16*)carve((size_t)DIM * DIM * 2);      // W_proj^T bf16
    __bf16* qkvb    = (__bf16*)carve((size_t)ROWS * QKV_N * 2);   // qkv bf16
    __bf16* attn_o  = (__bf16*)carve((size_t)ROWS * DIM * 2);     // attn out bf16
    (void)ws_size;

    // 1) conversions
    cvt_f32_to_bf16<<<1024, 256, 0, stream>>>(x, xb, ROWS * DIM);
    cvt_transpose_bf16<<<1024, 256, 0, stream>>>(W_qkv, wqkvT, DIM, QKV_N);
    cvt_transpose_bf16<<<512, 256, 0, stream>>>(W_proj, wprojT, DIM, DIM);

    // 2) qkv = x @ W_qkv   (12544 x 2304, K=768)
    gemm_wmma<true><<<dim3(QKV_N / BT_N, ROWS / BT_M), 256, 0, stream>>>(
        xb, wqkvT, nullptr, qkvb, ROWS, QKV_N, DIM);

    // 3) fused attention per (b,h)
    hipFuncSetAttribute(reinterpret_cast<const void*>(attn_kernel),
                        hipFuncAttributeMaxDynamicSharedMemorySize, LDS_TOTAL);
    attn_kernel<<<BATCH * HEADS, 256, LDS_TOTAL, stream>>>(qkvb, static_a, attn_o);

    // 4) out = attn_out @ W_proj + b_proj   (12544 x 768, K=768)
    gemm_wmma<false><<<dim3(DIM / BT_N, ROWS / BT_M), 256, 0, stream>>>(
        attn_o, wprojT, b_proj, out, ROWS, DIM, DIM);
}
